// GroupLinear_36971078484018
// MI455X (gfx1250) — matrix operand order repaired and checked by reading, not compiled, not run on hardware
//
#include <hip/hip_runtime.h>
#include <hip/hip_bf16.h>

typedef __attribute__((ext_vector_type(16))) __bf16 v16bf;
typedef __attribute__((ext_vector_type(8)))  float  v8f;

constexpr int kBatch  = 4096;
constexpr int kDIn    = 1024;
constexpr int kDOut   = 1024;
constexpr int kGroups = 16;

constexpr int BM = 128;   // block rows (batch)
constexpr int BN = 64;    // block cols (d_out)
constexpr int BK = 32;    // K step (WMMA bf16 K)
constexpr int KP = BK / 2;          // 16 k-pairs (dwords) per row
constexpr int LSTR = KP + 4;        // padded row stride in dwords (20): keeps 16B align,
                                    // 16 distinct banks over 16 fragment lanes

// Fragment: 16 bf16 per lane = two contiguous uint4 quads.
union FragBF { v16bf v; uint4 q[2]; };

// Split two fp32 into packed bf16 hi-pair (ret) and lo-pair (out).
// hi = trunc-bf16(f); lo = trunc-bf16(f - hi). Combined rel. error ~2^-16.
__device__ __forceinline__ unsigned split2(float a, float b, unsigned& lo) {
  unsigned ha = __float_as_uint(a) >> 16;
  unsigned hb = __float_as_uint(b) >> 16;
  float ra = a - __uint_as_float(ha << 16);
  float rb = b - __uint_as_float(hb << 16);
  lo = ((__float_as_uint(rb) >> 16) << 16) | (__float_as_uint(ra) >> 16);
  return (hb << 16) | ha;
}

// Convert 8 consecutive fp32 (two float4) -> 4 hi dwords + 4 lo dwords.
__device__ __forceinline__ void split8(const float4 v0, const float4 v1,
                                       uint4& hi, uint4& lo) {
  hi.x = split2(v0.x, v0.y, lo.x);
  hi.y = split2(v0.z, v0.w, lo.y);
  hi.z = split2(v1.x, v1.y, lo.z);
  hi.w = split2(v1.z, v1.w, lo.w);
}

__global__ __launch_bounds__(256, 2)
void GroupLinear_wmma_bf16x3(const float* __restrict__ X,
                             const float* __restrict__ W,
                             const float* __restrict__ Bias,
                             float* __restrict__ Out) {
  // Deinterleaved hi/lo bf16-pair tiles, row-major [row][k-pair], stride LSTR dwords.
  __shared__ unsigned lAhi[BM * LSTR];   // 10240 B
  __shared__ unsigned lAlo[BM * LSTR];   // 10240 B
  __shared__ unsigned lBhi[BN * LSTR];   //  5120 B
  __shared__ unsigned lBlo[BN * LSTR];   //  5120 B

  const int tid  = threadIdx.x;
  const int lane = tid & 31;
  const int wid  = tid >> 5;
  const int hl   = lane >> 4;     // lane half (0/1)
  const int l16  = lane & 15;

  const int n0 = blockIdx.x * BN;
  const int m0 = blockIdx.y * BM;
  const int g  = blockIdx.z;

  const int waveM = wid & 3;      // 4 waves along M: 32-row slices
  const int waveN = wid >> 2;     // 2 waves along N: 32-col slices

  const float* Wg = W + (size_t)g * kDOut * kDIn;

  v8f acc[2][2] = {};

  for (int kt = 0; kt < kDIn / BK; ++kt) {
    const int k0 = kt * BK;
    __syncthreads();

    // ---- stage X tile (128x32 fp32): 512 8-float chunks, 2 per thread ----
#pragma unroll
    for (int i = 0; i < 2; ++i) {
      const int id = i * 256 + tid;
      const int r  = id >> 2;          // 0..127
      const int oc = id & 3;           // 8-float chunk along K
      const float* src = X + (size_t)(m0 + r) * kDIn + k0 + oc * 8;
      const float4 v0 = *(const float4*)(src);
      const float4 v1 = *(const float4*)(src + 4);
      uint4 hi, lo;
      split8(v0, v1, hi, lo);
      const int d = r * LSTR + oc * 4;           // dword offset, 16B aligned
      *(uint4*)&lAhi[d] = hi;
      *(uint4*)&lAlo[d] = lo;
    }
    // ---- stage W tile (64x32 fp32): 256 8-float chunks, 1 per thread ----
    {
      const int r  = tid >> 2;         // 0..63 (relative output col)
      const int oc = tid & 3;
      const float* src = Wg + (size_t)(n0 + r) * kDIn + k0 + oc * 8;
      const float4 v0 = *(const float4*)(src);
      const float4 v1 = *(const float4*)(src + 4);
      uint4 hi, lo;
      split8(v0, v1, hi, lo);
      const int d = r * LSTR + oc * 4;
      *(uint4*)&lBhi[d] = hi;
      *(uint4*)&lBlo[d] = lo;
    }

    // prefetch next K tile (global_prefetch_b8)
    if (kt + 1 < kDIn / BK) {
      __builtin_prefetch(X  + (size_t)(m0 + (tid >> 1)) * kDIn + k0 + BK + (tid & 1) * 16, 0, 1);
      __builtin_prefetch(Wg + (size_t)(n0 + (tid & 63)) * kDIn + k0 + BK + (tid >> 7) * 16, 0, 1);
    }

    __syncthreads();

    // ---- fragments: contiguous ds_load_b128 into register quads ----
    FragBF aHi[2], aLo[2], bHi[2], bLo[2];
#pragma unroll
    for (int t = 0; t < 2; ++t) {
      // A 16x32 bf16 layout: quads at k-pairs [4*hl .. +3] and [8+4*hl .. +3]
      const int rA = waveM * 32 + t * 16 + l16;
      const int dA = rA * LSTR + 4 * hl;
      aHi[t].q[0] = *(const uint4*)&lAhi[dA];
      aHi[t].q[1] = *(const uint4*)&lAhi[dA + 8];
      aLo[t].q[0] = *(const uint4*)&lAlo[dA];
      aLo[t].q[1] = *(const uint4*)&lAlo[dA + 8];
      // B 32x16 bf16 layout: quads at k-pairs [4*hl .. +3] and [8+4*hl .. +3]
      const int rB = waveN * 32 + t * 16 + l16;
      const int dB = rB * LSTR + 4 * hl;
      bHi[t].q[0] = *(const uint4*)&lBhi[dB];
      bHi[t].q[1] = *(const uint4*)&lBhi[dB + 8];
      bLo[t].q[0] = *(const uint4*)&lBlo[dB];
      bLo[t].q[1] = *(const uint4*)&lBlo[dB + 8];
    }

    // ---- bf16x3 WMMA: hi*hi + lo*hi + hi*lo, fp32 accumulate ----
#pragma unroll
    for (int tm = 0; tm < 2; ++tm)
#pragma unroll
      for (int tn = 0; tn < 2; ++tn) {
        acc[tm][tn] = __builtin_amdgcn_wmma_f32_16x16x32_bf16(
            false, aHi[tm].v, false, bHi[tn].v, (short)0, acc[tm][tn], false, false);
        acc[tm][tn] = __builtin_amdgcn_wmma_f32_16x16x32_bf16(
            false, aLo[tm].v, false, bHi[tn].v, (short)0, acc[tm][tn], false, false);
        acc[tm][tn] = __builtin_amdgcn_wmma_f32_16x16x32_bf16(
            false, aHi[tm].v, false, bLo[tn].v, (short)0, acc[tm][tn], false, false);
      }
  }

  // ---- epilogue: bias add + store. C/D layout: VGPR v -> row v + 8*hl, col = l16 ----
#pragma unroll
  for (int tn = 0; tn < 2; ++tn) {
    const int o = n0 + waveN * 32 + tn * 16 + l16;
    const float bias = Bias[g * kDOut + o];
#pragma unroll
    for (int tm = 0; tm < 2; ++tm) {
      const int mBase = m0 + waveM * 32 + tm * 16 + hl * 8;
#pragma unroll
      for (int v = 0; v < 8; ++v) {
        Out[(size_t)(mBase + v) * (kGroups * kDOut) + (size_t)g * kDOut + o] =
            acc[tm][tn][v] + bias;
      }
    }
  }
}

extern "C" void kernel_launch(void* const* d_in, const int* in_sizes, int n_in,
                              void* d_out, int out_size, void* d_ws, size_t ws_size,
                              hipStream_t stream) {
  const float* X    = (const float*)d_in[0];
  const float* W    = (const float*)d_in[1];
  const float* Bias = (const float*)d_in[2];
  float* Out        = (float*)d_out;
  (void)in_sizes; (void)n_in; (void)out_size; (void)d_ws; (void)ws_size;

  dim3 grid(kDOut / BN, kBatch / BM, kGroups);   // (16, 32, 16)
  GroupLinear_wmma_bf16x3<<<grid, 256, 0, stream>>>(X, W, Bias, Out);
}
